// InstantNerf_54795192762577
// MI455X (gfx1250) — matrix operand order repaired and checked by reading, not compiled, not run on hardware
//
#include <hip/hip_runtime.h>
#include <hip/hip_bf16.h>
#include <math.h>

typedef __attribute__((ext_vector_type(16))) _Float16 v16h;
typedef __attribute__((ext_vector_type(8)))  _Float16 v8h;
typedef __attribute__((ext_vector_type(8)))  float    v8f;

#define NLEV 16
#define TBL  524288
#define TBLMASK (TBL - 1)
#define HTSTRIDE (NLEV * TBL)
#define NTILES 18

// ---- WMMA helpers (layouts per CDNA5 ISA) ----------------------

// Pre-swizzled B tile from LDS: lane's 16 halves are contiguous (32B aligned)
__device__ __forceinline__ v16h load_B_sw(const _Float16* __restrict__ s_w,
                                          int t, int lane) {
  return *(const v16h*)&s_w[(t * 32 + lane) * 16];
}

// A matrix 16x32 f16 from LDS row (row = &lds[m*stride]), K chunk at k0.
// Two contiguous 16B chunks -> ds_load_b128 pairs.
__device__ __forceinline__ v16h load_A_lds(const _Float16* __restrict__ row,
                                           int k0, int lane) {
  const int kb = (lane & 16) ? 8 : 0;
  const v8h lo = *(const v8h*)&row[k0 + kb];
  const v8h hi = *(const v8h*)&row[k0 + 16 + kb];
  v16h a;
#pragma unroll
  for (int e = 0; e < 8; ++e) { a[e] = lo[e]; a[8 + e] = hi[e]; }
  return a;
}

__device__ __forceinline__ v8f bias_init(const float* __restrict__ b, int n0, int lane) {
  const float v = b[n0 + (lane & 15)];
  v8f c;
#pragma unroll
  for (int r = 0; r < 8; ++r) c[r] = v;
  return c;
}

// C 16x16 f32 -> LDS f16 rows (optionally relu). Row = r + 8*(lane>=16), col = n0+lane%16
__device__ __forceinline__ void store_C_lds(_Float16* __restrict__ buf, int stride,
                                            int n0, v8f c, int lane, bool relu) {
  const int n  = n0 + (lane & 15);
  const int mb = (lane & 16) ? 8 : 0;
#pragma unroll
  for (int r = 0; r < 8; ++r) {
    float v = c[r];
    if (relu) v = fmaxf(v, 0.0f);
    buf[(mb + r) * stride + n] = (_Float16)v;
  }
}

#define WMMA(a, b, c) __builtin_amdgcn_wmma_f32_16x16x32_f16(false, (a), false, (b), (short)0, (c), false, false)

// ---------------------------------------------------------------------------

__global__ __launch_bounds__(256) void nerf_fused(
    const float* __restrict__ position, const float* __restrict__ direction,
    const float* __restrict__ htab,
    const float* __restrict__ w1,  const float* __restrict__ b1v,
    const float* __restrict__ wd,  const float* __restrict__ bdv,
    const float* __restrict__ wc1, const float* __restrict__ bc1v,
    const float* __restrict__ wc2, const float* __restrict__ bc2v,
    const float* __restrict__ wc3, const float* __restrict__ bc3v,
    float* __restrict__ out, int npts) {

  __shared__ float    s_frac[8][16][48];                 // flat j = level*3+dim
  __shared__ _Float16 s_x[8][16][32];                    // enc, later xcat
  __shared__ _Float16 s_a[8][16][64];                    // h, later c1
  __shared__ _Float16 s_b[8][16][64];                    // c2
  __shared__ __align__(32) _Float16 s_w[NTILES * 32 * 16]; // swizzled B tiles

  const int tid    = threadIdx.x;
  const int lane   = tid & 31;
  const int wv     = tid >> 5;
  const int base   = (blockIdx.x * 8 + wv) * 16; // 16 points per wave
  const int pLocal = lane & 15;
  const int p      = base + pLocal;

  const float* ht0 = htab;
  const float* ht1 = htab + (size_t)HTSTRIDE;
  float* outDen = out;
  float* outCol = out + (size_t)npts * 16;

  // ---- cooperative fill of swizzled f16 weight tiles (B layout) ----------
  // tiles: 0-3 w1(k0=0,n=16t) | 4-5 wd(k0=0/32) | 6-9 wc1 | 10-13 wc2 k0=0 | 14-17 wc2 k0=32
  {
    const float* Wt[NTILES];
    int ldt[NTILES], k0t[NTILES], n0t[NTILES];
#pragma unroll
    for (int t = 0; t < 4; ++t)  { Wt[t]      = w1;  ldt[t]      = 64; k0t[t]      = 0;  n0t[t]      = 16 * t; }
    Wt[4] = wd;  ldt[4] = 16; k0t[4] = 0;  n0t[4] = 0;
    Wt[5] = wd;  ldt[5] = 16; k0t[5] = 32; n0t[5] = 0;
#pragma unroll
    for (int t = 0; t < 4; ++t)  { Wt[6 + t]  = wc1; ldt[6 + t]  = 64; k0t[6 + t]  = 0;  n0t[6 + t]  = 16 * t; }
#pragma unroll
    for (int t = 0; t < 4; ++t)  { Wt[10 + t] = wc2; ldt[10 + t] = 64; k0t[10 + t] = 0;  n0t[10 + t] = 16 * t; }
#pragma unroll
    for (int t = 0; t < 4; ++t)  { Wt[14 + t] = wc2; ldt[14 + t] = 64; k0t[14 + t] = 32; n0t[14 + t] = 16 * t; }

    for (int i = tid; i < NTILES * 32 * 16; i += 256) {
      const int t    = i >> 9;          // /512
      const int r    = i & 511;
      const int slot = r >> 4;          // lane slot 0..31
      const int e    = r & 15;          // half index
      const int k    = k0t[t] + ((slot & 16) ? 8 : 0) + ((e < 8) ? e : 8 + e);
      const int n    = n0t[t] + (slot & 15);
      s_w[i] = (_Float16)Wt[t][k * ldt[t] + n];
    }
  }

  // ---- hash encode: lane handles point pLocal, levels lgrp..lgrp+7 -------
  const float px0 = position[(size_t)p * 3 + 0];
  const float py0 = position[(size_t)p * 3 + 1];
  const float pz0 = position[(size_t)p * 3 + 2];
  const int lgrp = (lane & 16) ? 8 : 0;

  // pass 1: fractions (stored flat to replicate reference reshape quirk)
#pragma unroll
  for (int li = 0; li < 8; ++li) {
    const int L = lgrp + li;
    const float s = floorf(16.0f * exp2f(0.4f * (float)L)); // 64^(L/15)
    const float sx = px0 * s, sy = py0 * s, sz = pz0 * s;
    s_frac[wv][pLocal][L * 3 + 0] = sx - floorf(sx);
    s_frac[wv][pLocal][L * 3 + 1] = sy - floorf(sy);
    s_frac[wv][pLocal][L * 3 + 2] = sz - floorf(sz);
  }
  __syncthreads();

  // pass 2: gather 8 corners * 2 feats, trilerp with scrambled weights
#pragma unroll 1
  for (int li = 0; li < 8; ++li) {
    const int L = lgrp + li;
    const float s = floorf(16.0f * exp2f(0.4f * (float)L));
    const float sx = px0 * s, sy = py0 * s, sz = pz0 * s;
    const unsigned fx = (unsigned)(int)floorf(sx), cx = (unsigned)(int)ceilf(sx);
    const unsigned fy = (unsigned)(int)floorf(sy), cy = (unsigned)(int)ceilf(sy);
    const unsigned fz = (unsigned)(int)floorf(sz), cz = (unsigned)(int)ceilf(sz);
    const unsigned fxp = fx,               cxp = cx;
    const unsigned fyp = fy * 2654435761u, cyp = cy * 2654435761u;
    const unsigned fzp = fz * 805459861u,  czp = cz * 805459861u;
    const int lb = L * TBL;
    const int i0 = (int)((cxp ^ cyp ^ czp) & TBLMASK) + lb; // (1,1,1)
    const int i1 = (int)((cxp ^ cyp ^ fzp) & TBLMASK) + lb; // (1,1,0)
    const int i2 = (int)((cxp ^ fyp ^ czp) & TBLMASK) + lb; // (1,0,1)
    const int i3 = (int)((fxp ^ cyp ^ czp) & TBLMASK) + lb; // (0,1,1)
    const int i4 = (int)((cxp ^ fyp ^ fzp) & TBLMASK) + lb; // (1,0,0)
    const int i5 = (int)((fxp ^ cyp ^ fzp) & TBLMASK) + lb; // (0,1,0)
    const int i6 = (int)((fxp ^ fyp ^ czp) & TBLMASK) + lb; // (0,0,1)
    const int i7 = (int)((fxp ^ fyp ^ fzp) & TBLMASK) + lb; // (0,0,0)

    const float wx = s_frac[wv][pLocal][L];
    const float wy = s_frac[wv][pLocal][16 + L];
    const float wz = s_frac[wv][pLocal][32 + L];

#pragma unroll
    for (int k = 0; k < 2; ++k) {
      const float* ht = k ? ht1 : ht0;
      const float f0 = ht[i0], f1 = ht[i1], f2 = ht[i2], f3 = ht[i3];
      const float f4 = ht[i4], f5 = ht[i5], f6 = ht[i6], f7 = ht[i7];
      const float f03 = f0 * wx + f3 * (1.0f - wx);
      const float f12 = f1 * wx + f2 * (1.0f - wx);
      const float f56 = f5 * wx + f6 * (1.0f - wx);
      const float f47 = f4 * wx + f7 * (1.0f - wx);
      const float e = (f03 * wy + f12 * (1.0f - wy)) * wz +
                      (f47 * wy + f56 * (1.0f - wy)) * (1.0f - wz);
      s_x[wv][pLocal][L * 2 + k] = (_Float16)e;
    }
  }
  __syncthreads();

  // ---- layer 1: h = relu(enc @ w1 + b1), K=32, N=64 ----------------------
  {
    const v16h a = load_A_lds(&s_x[wv][pLocal][0], 0, lane);
#pragma unroll
    for (int t = 0; t < 4; ++t) {
      v8f c = bias_init(b1v, 16 * t, lane);
      c = WMMA(a, load_B_sw(s_w, t, lane), c);
      store_C_lds(&s_a[wv][0][0], 64, 16 * t, c, lane, true);
    }
  }
  __syncthreads();

  // ---- density = h @ wd + bd, K=64, N=16 ---------------------------------
  {
    const v16h a0 = load_A_lds(&s_a[wv][pLocal][0], 0, lane);
    const v16h a1 = load_A_lds(&s_a[wv][pLocal][0], 32, lane);
    v8f c = bias_init(bdv, 0, lane);
    c = WMMA(a0, load_B_sw(s_w, 4, lane), c);
    c = WMMA(a1, load_B_sw(s_w, 5, lane), c);
    const int n  = lane & 15;
    const int mb = (lane & 16) ? 8 : 0;
#pragma unroll
    for (int r = 0; r < 8; ++r) {
      const int m  = mb + r;
      const int pt = base + m;
      outDen[(size_t)pt * 16 + n] = c[r];        // raw density out
      s_x[wv][m][n] = (_Float16)c[r];            // xcat[:, 0:16] (no relu)
    }
  }

  // ---- SH encode into xcat[:, 16:32] -------------------------------------
  if (lane < 16) {
    const int pt = base + lane;
    const float x = direction[(size_t)pt * 3 + 0];
    const float y = direction[(size_t)pt * 3 + 1];
    const float z = direction[(size_t)pt * 3 + 2];
    const float xx = x * x, yy = y * y, zz = z * z;
    float sh[16];
    sh[0]  = 0.28209479177387814f;
    sh[1]  = 0.4886025119029199f * y;
    sh[2]  = 0.4886025119029199f * z;
    sh[3]  = 0.4886025119029199f * x;
    sh[4]  = 1.0925484305920792f * x * y;
    sh[5]  = 1.0925484305920792f * y * z;
    sh[6]  = 0.9461746957575601f * zz - 0.31539156525252f;
    sh[7]  = 1.0925484305920792f * x * z;
    sh[8]  = 0.5462742152960396f * (xx - yy);
    sh[9]  = 0.5900435899266435f * y * (3.0f * xx - yy);
    sh[10] = 2.890611442640554f * x * y * z;
    sh[11] = 0.4570457994644658f * y * (5.0f * zz - 1.0f);
    sh[12] = 0.3731763325901154f * z * (5.0f * zz - 3.0f);
    sh[13] = 0.4570457994644658f * x * (5.0f * zz - 1.0f);
    sh[14] = 1.445305721320277f * z * (xx - yy);
    sh[15] = 0.5900435899266435f * x * (xx - 3.0f * yy);
#pragma unroll
    for (int j = 0; j < 16; ++j) s_x[wv][lane][16 + j] = (_Float16)sh[j];
  }
  __syncthreads();

  // ---- c1 = relu(xcat @ wc1 + bc1), K=32, N=64 ---------------------------
  {
    const v16h a = load_A_lds(&s_x[wv][pLocal][0], 0, lane);
#pragma unroll
    for (int t = 0; t < 4; ++t) {
      v8f c = bias_init(bc1v, 16 * t, lane);
      c = WMMA(a, load_B_sw(s_w, 6 + t, lane), c);
      store_C_lds(&s_a[wv][0][0], 64, 16 * t, c, lane, true);
    }
  }
  __syncthreads();

  // ---- c2 = relu(c1 @ wc2 + bc2), K=64, N=64 -----------------------------
  {
    const v16h a0 = load_A_lds(&s_a[wv][pLocal][0], 0, lane);
    const v16h a1 = load_A_lds(&s_a[wv][pLocal][0], 32, lane);
#pragma unroll
    for (int t = 0; t < 4; ++t) {
      v8f c = bias_init(bc2v, 16 * t, lane);
      c = WMMA(a0, load_B_sw(s_w, 10 + t, lane), c);
      c = WMMA(a1, load_B_sw(s_w, 14 + t, lane), c);
      store_C_lds(&s_b[wv][0][0], 64, 16 * t, c, lane, true);
    }
  }
  __syncthreads();

  // ---- color = sigmoid(c2 @ wc3 + bc3), N=3 (scalar tail) ----------------
  if (lane < 16) {
    const int pt = base + lane;
    float a0 = bc3v[0], a1 = bc3v[1], a2 = bc3v[2];
#pragma unroll
    for (int k = 0; k < 64; ++k) {
      const float v = (float)s_b[wv][lane][k];
      a0 += v * wc3[k * 3 + 0];
      a1 += v * wc3[k * 3 + 1];
      a2 += v * wc3[k * 3 + 2];
    }
    outCol[(size_t)pt * 3 + 0] = 1.0f / (1.0f + expf(-a0));
    outCol[(size_t)pt * 3 + 1] = 1.0f / (1.0f + expf(-a1));
    outCol[(size_t)pt * 3 + 2] = 1.0f / (1.0f + expf(-a2));
  }
}

extern "C" void kernel_launch(void* const* d_in, const int* in_sizes, int n_in,
                              void* d_out, int out_size, void* d_ws, size_t ws_size,
                              hipStream_t stream) {
  const float* position  = (const float*)d_in[0];
  const float* direction = (const float*)d_in[1];
  const float* htab      = (const float*)d_in[2];
  const float* w1  = (const float*)d_in[3];
  const float* b1  = (const float*)d_in[4];
  const float* wd  = (const float*)d_in[5];
  const float* bd  = (const float*)d_in[6];
  const float* wc1 = (const float*)d_in[7];
  const float* bc1 = (const float*)d_in[8];
  const float* wc2 = (const float*)d_in[9];
  const float* bc2 = (const float*)d_in[10];
  const float* wc3 = (const float*)d_in[11];
  const float* bc3 = (const float*)d_in[12];

  const int npts = in_sizes[0] / 3;          // 262144
  const int grid = npts / 128;               // 8 waves * 16 points per block

  nerf_fused<<<grid, 256, 0, stream>>>(position, direction, htab,
                                       w1, b1, wd, bd, wc1, bc1, wc2, bc2, wc3, bc3,
                                       (float*)d_out, npts);
}
